// head_attention_mixed_12034498363712
// MI455X (gfx1250) — matrix operand order repaired and checked by reading, not compiled, not run on hardware
//
#include <hip/hip_runtime.h>
#include <hip/hip_bf16.h>

#define NN 256
#define DD 256
#define DHH 64

typedef __attribute__((ext_vector_type(16))) __bf16    v16bf;
typedef __attribute__((ext_vector_type(8)))  float     v8f;
typedef __attribute__((ext_vector_type(8)))  unsigned  v8u;

__device__ __forceinline__ unsigned short f2bf(float f) {
    unsigned u = __builtin_bit_cast(unsigned, f);
    u += 0x7FFFu + ((u >> 16) & 1u);            // round-to-nearest-even
    return (unsigned short)(u >> 16);
}

// Pack two f32 -> one dword of bf16 (low = lo) with round-half-up:
// one v_add each + one v_perm_b32 to gather the two high halves.
__device__ __forceinline__ unsigned pk2bf(float lo, float hi) {
    unsigned ul = __builtin_bit_cast(unsigned, lo) + 0x8000u;
    unsigned uh = __builtin_bit_cast(unsigned, hi) + 0x8000u;
    return __builtin_amdgcn_perm(uh, ul, 0x07060302u);  // {uh[31:16], ul[31:16]}
}

// Build a v16bf B-fragment from two 16B-aligned bf16 LDS locations.
__device__ __forceinline__ v16bf ld_frag_bf16(const unsigned short* p0, const unsigned short* p1) {
    uint4 a = *(const uint4*)p0;
    uint4 b = *(const uint4*)p1;
    v8u u;
    u[0]=a.x; u[1]=a.y; u[2]=a.z; u[3]=a.w;
    u[4]=b.x; u[5]=b.y; u[6]=b.z; u[7]=b.w;
    return __builtin_bit_cast(v16bf, u);
}

// Build a v16bf A-fragment from f32 LDS (p = row base + 8*half floats):
// elements 0..7  <- K = 8*half+0..7    (floats p[0..7])
// elements 8..15 <- K = 16+8*half+0..7 (floats p[16..23])
__device__ __forceinline__ v16bf ld_frag_f32(const float* p) {
    float4 a = *(const float4*)(p);
    float4 b = *(const float4*)(p + 4);
    float4 c = *(const float4*)(p + 16);
    float4 d = *(const float4*)(p + 20);
    v8u u;
    u[0] = pk2bf(a.x, a.y); u[1] = pk2bf(a.z, a.w);
    u[2] = pk2bf(b.x, b.y); u[3] = pk2bf(b.z, b.w);
    u[4] = pk2bf(c.x, c.y); u[5] = pk2bf(c.z, c.w);
    u[6] = pk2bf(d.x, d.y); u[7] = pk2bf(d.z, d.w);
    return __builtin_bit_cast(v16bf, u);
}

// ---------------------------------------------------------------------------
// Kernel A: six x-projections (Qn=n2e_q, Kn=e2n_k, Vn=e2n_v, Qs/Ks/Vs=n2n_*)
// ---------------------------------------------------------------------------
__global__ void __launch_bounds__(64) node_proj_kernel(
    const float* __restrict__ x, const float* __restrict__ nmask,
    const float* __restrict__ W0, const float* __restrict__ B0,
    const float* __restrict__ W1, const float* __restrict__ B1,
    const float* __restrict__ W2, const float* __restrict__ B2,
    const float* __restrict__ W3, const float* __restrict__ B3,
    const float* __restrict__ W4, const float* __restrict__ B4,
    const float* __restrict__ W5, const float* __restrict__ B5,
    float* __restrict__ O0, float* __restrict__ O1, float* __restrict__ O2,
    float* __restrict__ O3, float* __restrict__ O4, float* __restrict__ O5)
{
    __shared__ float xrow[DD];
    const int bi = blockIdx.x;
    const int t  = threadIdx.x;
    #pragma unroll
    for (int q = 0; q < 4; ++q)
        xrow[t + 64*q] = x[(size_t)bi*DD + t + 64*q];
    __syncthreads();

    const float nm = nmask[bi];
    const float* Ws[6] = {W0,W1,W2,W3,W4,W5};
    const float* Bs[6] = {B0,B1,B2,B3,B4,B5};
    float*       Os[6] = {O0,O1,O2,O3,O4,O5};
    #pragma unroll
    for (int w = 0; w < 6; ++w) {
        float s = Bs[w][t];
        const float* Wp = Ws[w];
        #pragma unroll 8
        for (int k = 0; k < DD; ++k) s += xrow[k] * Wp[k*DHH + t];
        Os[w][(size_t)bi*DHH + t] = s * nm;
    }
}

// ---------------------------------------------------------------------------
// Kernel B: fused edge pipeline, one workgroup (8 waves) per (b,i).
//   e chunks streamed via global_load_async_to_lds_b128 (double buffered),
//   Ke/Ve/Qe via v_wmma_f32_16x16x32_bf16, then softmax / x_cross / e_new.
// LDS: sW[3][64][264] bf16 + sEf[2][256][36] f32 + buffers  (~172 KB)
// ---------------------------------------------------------------------------
#define SW_STRIDE 264
#define SE_STRIDE 36
#define SW_BYTES  (3*64*SW_STRIDE*2)                 // 101376
#define SE_BYTES  (2*256*SE_STRIDE*4)                // 73728
#define SMEM_B    (SW_BYTES + SE_BYTES + (256+32+64)*4)

__global__ void __launch_bounds__(256) fused_edge_kernel(
    const float* __restrict__ e,  const float* __restrict__ nmask,
    const float* __restrict__ Wk, const float* __restrict__ bk,
    const float* __restrict__ Wv, const float* __restrict__ bv,
    const float* __restrict__ Wq, const float* __restrict__ bq,
    const float* __restrict__ Qn, const float* __restrict__ Kn,
    const float* __restrict__ Vn,
    float* __restrict__ xc_ws, float* __restrict__ e_out)
{
    extern __shared__ char smem[];
    unsigned short* sW   = (unsigned short*)smem;            // 3*64*264 bf16
    float*          sEf  = (float*)(smem + SW_BYTES);        // 2*256*36 f32
    float*          attb = (float*)(smem + SW_BYTES + SE_BYTES);
    float*          red  = attb + 256;
    float*          xcb  = red + 32;

    const int t     = threadIdx.x;
    const int lane  = t & 31, wv = t >> 5;
    const int half  = lane >> 4, colc = lane & 15;
    const int bi    = blockIdx.x;          // b*N + i
    const int bnode = bi & ~255;           // b*N
    const int i     = bi & 255;

    // ---- stage weights transposed to bf16 LDS: sW[w][n][k] ----
    const float* Wsrc[3] = {Wk, Wv, Wq};
    #pragma unroll
    for (int w = 0; w < 3; ++w) {
        const float* Wp = Wsrc[w];
        for (int it = 0; it < 64; ++it) {
            int idx = t + it*256;                 // 256*64 elements
            int k = idx >> 6, n = idx & 63;
            sW[(w*64 + n)*SW_STRIDE + k] = f2bf(Wp[idx]);
        }
    }

    // async e-chunk stager: each lane copies 8x 16B, lane->(jj,k4)
    const unsigned long long gbase =
        (unsigned long long)(const void*)(e + (size_t)bi * NN * DD);
    const int k4 = t & 7, jr = t >> 3;
    auto issue_chunk = [&](int kc, int buf) {
        #pragma unroll
        for (int it = 0; it < 8; ++it) {
            int jj = it*32 + jr;
            unsigned goff = (unsigned)((jj*DD + kc*32 + k4*4) * 4);
            unsigned lds  = (unsigned)(size_t)(const void*)
                            (sEf + (buf*256 + jj)*SE_STRIDE + k4*4);
            asm volatile("global_load_async_to_lds_b128 %0, %1, %2"
                         :: "v"(lds), "v"(goff), "s"(gbase) : "memory");
        }
    };
    issue_chunk(0, 0);

    v8f acc[2][3][4];                             // [mtile][Ke/Ve/Qe][ntile]
    #pragma unroll
    for (int m = 0; m < 2; ++m)
        #pragma unroll
        for (int w = 0; w < 3; ++w)
            #pragma unroll
            for (int nt = 0; nt < 4; ++nt)
                acc[m][w][nt] = (v8f){0.f,0.f,0.f,0.f,0.f,0.f,0.f,0.f};

    // ---- K-chunk loop: 8 chunks of 32 over D=256, double buffered ----
    for (int kc = 0; kc < 8; ++kc) {
        asm volatile("s_wait_asynccnt 0x0" ::: "memory");
        __syncthreads();
        if (kc < 7) issue_chunk(kc + 1, (kc + 1) & 1);

        // B fragments: W^T rows contiguous in k -> 2x ds_load_b128 each
        v16bf bfr[3][4];
        #pragma unroll
        for (int w = 0; w < 3; ++w)
            #pragma unroll
            for (int nt = 0; nt < 4; ++nt) {
                const unsigned short* p =
                    sW + (w*64 + colc + 16*nt)*SW_STRIDE + kc*32 + 8*half;
                bfr[w][nt] = ld_frag_bf16(p, p + 16);
            }

        const float* ebuf = sEf + (kc & 1)*256*SE_STRIDE;
        #pragma unroll
        for (int m = 0; m < 2; ++m) {
            int row = wv*32 + m*16 + colc;        // j row of this lane
            v16bf af = ld_frag_f32(ebuf + row*SE_STRIDE + 8*half);
            #pragma unroll
            for (int w = 0; w < 3; ++w)
                #pragma unroll
                for (int nt = 0; nt < 4; ++nt)
                    acc[m][w][nt] = __builtin_amdgcn_wmma_f32_16x16x32_bf16(
                        false, af, false, bfr[w][nt], (short)0, acc[m][w][nt],
                        false, false);
        }
    }

    // ---- bias + edge mask (em1*em2) ----
    const float* bptr[3] = {bk, bv, bq};
    float bwv[3][4];
    #pragma unroll
    for (int w = 0; w < 3; ++w)
        #pragma unroll
        for (int nt = 0; nt < 4; ++nt) bwv[w][nt] = bptr[w][colc + 16*nt];
    const float nmI = nmask[bnode + i];
    #pragma unroll
    for (int m = 0; m < 2; ++m)
        #pragma unroll
        for (int r = 0; r < 8; ++r) {
            int j = wv*32 + m*16 + r + 8*half;
            float em = nmI * nmask[bnode + j];
            #pragma unroll
            for (int w = 0; w < 3; ++w)
                #pragma unroll
                for (int nt = 0; nt < 4; ++nt)
                    acc[m][w][nt][r] = (acc[m][w][nt][r] + bwv[w][nt]) * em;
        }

    // ---- n2e logits: dot(Qn[b,i], Ke[j,:]) over n (lanes hold columns) ----
    float qn[4];
    #pragma unroll
    for (int nt = 0; nt < 4; ++nt) qn[nt] = Qn[(size_t)bi*DHH + colc + 16*nt];
    #pragma unroll
    for (int m = 0; m < 2; ++m)
        #pragma unroll
        for (int r = 0; r < 8; ++r) {
            float s = 0.f;
            #pragma unroll
            for (int nt = 0; nt < 4; ++nt) s += acc[m][0][nt][r] * qn[nt];
            s += __shfl_xor(s, 1); s += __shfl_xor(s, 2);
            s += __shfl_xor(s, 4); s += __shfl_xor(s, 8);
            if (colc == 0) {
                int j = wv*32 + m*16 + r + 8*half;
                float em = nmI * nmask[bnode + j];
                attb[j] = (em > 0.f) ? s * 0.125f : -1.0e9f;
            }
        }
    __syncthreads();

    // ---- softmax over j (two-stage block reduce) ----
    {
        float v = attb[t];
        float wm = v;
        #pragma unroll
        for (int msk = 1; msk < 32; msk <<= 1) wm = fmaxf(wm, __shfl_xor(wm, msk));
        if (lane == 0) red[wv] = wm;
        __syncthreads();
        float g = red[0];
        #pragma unroll
        for (int k = 1; k < 8; ++k) g = fmaxf(g, red[k]);
        float p = __expf(v - g);
        float sm = p;
        #pragma unroll
        for (int msk = 1; msk < 32; msk <<= 1) sm += __shfl_xor(sm, msk);
        if (lane == 0) red[8 + wv] = sm;
        __syncthreads();
        float gs = 0.f;
        #pragma unroll
        for (int k = 0; k < 8; ++k) gs += red[8 + k];
        attb[t] = p / gs;
        if (t < 64) xcb[t] = 0.f;
    }
    __syncthreads();

    // ---- x_cross = sum_j p[j] * Ve[j,:]  (per-column LDS atomics) ----
    #pragma unroll
    for (int m = 0; m < 2; ++m)
        #pragma unroll
        for (int nt = 0; nt < 4; ++nt) {
            float a = 0.f;
            #pragma unroll
            for (int r = 0; r < 8; ++r) {
                int j = wv*32 + m*16 + r + 8*half;
                a += attb[j] * acc[m][1][nt][r];
            }
            atomicAdd(&xcb[colc + 16*nt], a);
        }
    __syncthreads();
    if (t < 64) xc_ws[(size_t)bi*DHH + t] = xcb[t];

    // ---- e2n path: att_i/att_j -> e_new ----
    float kni[4], vni[4];
    #pragma unroll
    for (int nt = 0; nt < 4; ++nt) {
        kni[nt] = Kn[(size_t)bi*DHH + colc + 16*nt];
        vni[nt] = Vn[(size_t)bi*DHH + colc + 16*nt];
    }
    #pragma unroll
    for (int m = 0; m < 2; ++m)
        #pragma unroll
        for (int r = 0; r < 8; ++r) {
            int j = wv*32 + m*16 + r + 8*half;
            const float* knj = Kn + ((size_t)bnode + j)*DHH;
            float si = 0.f, sj = 0.f;
            #pragma unroll
            for (int nt = 0; nt < 4; ++nt) {
                float q = acc[m][2][nt][r];
                si += q * knj[colc + 16*nt];
                sj += q * kni[nt];
            }
            si += __shfl_xor(si, 1); si += __shfl_xor(si, 2);
            si += __shfl_xor(si, 4); si += __shfl_xor(si, 8);
            sj += __shfl_xor(sj, 1); sj += __shfl_xor(sj, 2);
            sj += __shfl_xor(sj, 4); sj += __shfl_xor(sj, 8);
            float ai = __expf(si * 0.125f);
            float aj = __expf(sj * 0.125f);
            float inv = 1.0f / (ai + aj);
            float em  = nmI * nmask[bnode + j];
            const float* vnj = Vn + ((size_t)bnode + j)*DHH;
            float* op = e_out + ((size_t)bi*NN + j)*DHH;
            #pragma unroll
            for (int nt = 0; nt < 4; ++nt) {
                int n = colc + 16*nt;
                op[n] = (ai*inv*vnj[n] + aj*inv*vni[nt]) * em;
            }
        }
}

// ---------------------------------------------------------------------------
// Kernel C: n2n self-attention + output mix. One block per batch b.
// ---------------------------------------------------------------------------
__global__ void __launch_bounds__(256) self_mix_kernel(
    const float* __restrict__ nmask,
    const float* __restrict__ Qs, const float* __restrict__ Ks,
    const float* __restrict__ Vs, const float* __restrict__ xc,
    const float* __restrict__ Wm, const float* __restrict__ bm,
    float* __restrict__ xout)
{
    extern __shared__ float sKV[];
    float* sK = sKV;
    float* sV = sKV + NN*DHH;
    const int b = blockIdx.x, t = threadIdx.x;
    for (int it = 0; it < 64; ++it) {
        int idx = t + it*256;
        sK[idx] = Ks[(size_t)b*NN*DHH + idx];
        sV[idx] = Vs[(size_t)b*NN*DHH + idx];
    }
    __syncthreads();

    const int i = t;
    const float nmI = nmask[b*NN + i];
    float q[DHH];
    #pragma unroll
    for (int h = 0; h < DHH; ++h) q[h] = Qs[((size_t)b*NN + i)*DHH + h];

    float mx = -1.0e30f;
    for (int j = 0; j < NN; ++j) {
        float s = 0.f;
        #pragma unroll
        for (int h = 0; h < DHH; ++h) s += q[h] * sK[j*DHH + h];
        float em = nmI * nmask[b*NN + j];
        float sc = (em > 0.f) ? s * 0.125f : -1.0e9f;
        mx = fmaxf(mx, sc);
    }
    float accv[DHH];
    #pragma unroll
    for (int h = 0; h < DHH; ++h) accv[h] = 0.f;
    float sum = 0.f;
    for (int j = 0; j < NN; ++j) {
        float s = 0.f;
        #pragma unroll
        for (int h = 0; h < DHH; ++h) s += q[h] * sK[j*DHH + h];
        float em = nmI * nmask[b*NN + j];
        float sc = (em > 0.f) ? s * 0.125f : -1.0e9f;
        float p = __expf(sc - mx);
        sum += p;
        #pragma unroll
        for (int h = 0; h < DHH; ++h) accv[h] += p * sV[j*DHH + h];
    }
    const float inv = nmI / sum;

    float xcv[DHH];
    #pragma unroll
    for (int h = 0; h < DHH; ++h) xcv[h] = xc[((size_t)b*NN + i)*DHH + h];

    for (int n = 0; n < DHH; ++n) {
        float o = bm[n];
        #pragma unroll
        for (int h = 0; h < DHH; ++h)
            o += xcv[h]*Wm[h*DHH + n] + (accv[h]*inv)*Wm[(DHH + h)*DHH + n];
        xout[((size_t)b*NN + i)*DHH + n] = o;
    }
}

// ---------------------------------------------------------------------------
extern "C" void kernel_launch(void* const* d_in, const int* in_sizes, int n_in,
                              void* d_out, int out_size, void* d_ws, size_t ws_size,
                              hipStream_t stream) {
    (void)in_sizes; (void)n_in; (void)out_size; (void)ws_size;
    const float* x     = (const float*)d_in[0];
    const float* e     = (const float*)d_in[1];
    const float* nmask = (const float*)d_in[2];
    const float* W_n2e_q = (const float*)d_in[3],  *b_n2e_q = (const float*)d_in[4];
    const float* W_n2e_k = (const float*)d_in[5],  *b_n2e_k = (const float*)d_in[6];
    const float* W_n2e_v = (const float*)d_in[7],  *b_n2e_v = (const float*)d_in[8];
    const float* W_e2n_q = (const float*)d_in[9],  *b_e2n_q = (const float*)d_in[10];
    const float* W_e2n_k = (const float*)d_in[11], *b_e2n_k = (const float*)d_in[12];
    const float* W_e2n_v = (const float*)d_in[13], *b_e2n_v = (const float*)d_in[14];
    const float* W_n2n_q = (const float*)d_in[15], *b_n2n_q = (const float*)d_in[16];
    const float* W_n2n_k = (const float*)d_in[17], *b_n2n_k = (const float*)d_in[18];
    const float* W_n2n_v = (const float*)d_in[19], *b_n2n_v = (const float*)d_in[20];
    const float* W_mix   = (const float*)d_in[21], *b_mix   = (const float*)d_in[22];

    const size_t SZ = (size_t)8 * NN * DHH;      // 131072
    float* ws = (float*)d_ws;
    float* Qn = ws;           float* Kn  = ws + SZ;   float* Vn  = ws + 2*SZ;
    float* Qs = ws + 3*SZ;    float* Ksv = ws + 4*SZ; float* Vsv = ws + 5*SZ;
    float* xc = ws + 6*SZ;

    float* x_out = (float*)d_out;                // (8,256,64)
    float* e_new = (float*)d_out + SZ;           // (8,256,256,64)

    node_proj_kernel<<<8*NN, 64, 0, stream>>>(
        x, nmask,
        W_n2e_q, b_n2e_q, W_e2n_k, b_e2n_k, W_e2n_v, b_e2n_v,
        W_n2n_q, b_n2n_q, W_n2n_k, b_n2n_k, W_n2n_v, b_n2n_v,
        Qn, Kn, Vn, Qs, Ksv, Vsv);

    fused_edge_kernel<<<8*NN, 256, SMEM_B, stream>>>(
        e, nmask,
        W_n2e_k, b_n2e_k, W_n2e_v, b_n2e_v, W_e2n_q, b_e2n_q,
        Qn, Kn, Vn, xc, e_new);

    self_mix_kernel<<<8, 256, 2*NN*DHH*sizeof(float), stream>>>(
        nmask, Qs, Ksv, Vsv, xc, W_mix, b_mix, x_out);
}
